// Attention_24704651887167
// MI455X (gfx1250) — matrix operand order repaired and checked by reading, not compiled, not run on hardware
//
#include <hip/hip_runtime.h>

typedef __attribute__((ext_vector_type(16))) _Float16 v16h;
typedef __attribute__((ext_vector_type(8)))  _Float16 v8h;
typedef __attribute__((ext_vector_type(4)))  _Float16 v4h;
typedef __attribute__((ext_vector_type(8)))  float    v8f;
typedef __attribute__((ext_vector_type(4)))  int      v4i;

#define WMMA(a, b, c) __builtin_amdgcn_wmma_f32_16x16x32_f16( \
    false, (a), false, (b), (short)0, (c), false, false)

constexpr int Bb = 8, Nn = 1024, Cc = 768, Hh = 12, Dd = 64;
constexpr int C3 = 3 * Cc;
constexpr float SCALE = 0.125f;  // 64^-0.5

// ---------------------------------------------------------------- async copy helpers
#if __has_builtin(__builtin_amdgcn_global_load_async_to_lds_b128)
#define ASYNC_BUILTIN 1
#else
#define ASYNC_BUILTIN 0
#endif

__device__ __forceinline__ void async_copy_b128(const _Float16* g, _Float16* l) {
#if ASYNC_BUILTIN
  // Builtin signature (from compiler diagnostic): first arg is
  // 'int __vector_size__(16) addrspace(1)*' (global), second is the LDS dest.
  __builtin_amdgcn_global_load_async_to_lds_b128(
      (__attribute__((address_space(1))) v4i*)g,
      (__attribute__((address_space(3))) v4i*)l, 0, 0);
#else
  asm volatile("global_load_async_to_lds_b128 %0, %1, off"
               :
               : "v"((unsigned)(uintptr_t)l), "v"((unsigned long long)(uintptr_t)g)
               : "memory");
#endif
}

__device__ __forceinline__ void wait_async0() {
#if __has_builtin(__builtin_amdgcn_s_wait_asynccnt)
  __builtin_amdgcn_s_wait_asynccnt(0);
#else
  asm volatile("s_wait_asynccnt 0x0" ::: "memory");
#endif
}

// ---------------------------------------------------------------- fp32 -> fp16
__global__ void __launch_bounds__(256) cvt_f32_f16(const float* __restrict__ src,
                                                   _Float16* __restrict__ dst) {
  size_t i = ((size_t)blockIdx.x * 256 + threadIdx.x) * 4;
  float4 f = *(const float4*)(src + i);
  v4h h = {(_Float16)f.x, (_Float16)f.y, (_Float16)f.z, (_Float16)f.w};
  *(v4h*)(dst + i) = h;
}

// ---------------------------------------------------------------- QKV GEMM
// Out[row, c3] = sum_k X[row,k] * Wqkv[c3,k];  scattered into q/k/v [B,H,N,D] f16
__global__ void __launch_bounds__(256) qkv_gemm(const _Float16* __restrict__ X,
                                                const _Float16* __restrict__ W,
                                                _Float16* __restrict__ Q,
                                                _Float16* __restrict__ K,
                                                _Float16* __restrict__ V) {
  const int tid = threadIdx.x;
  const int w = tid >> 5, lane = tid & 31;
  const int nn = lane & 15, hi = lane >> 4;
  const int row0 = blockIdx.y * 128 + w * 16;
  const int col0 = blockIdx.x * 64;
  const int koff = hi * 16;   // A-fragment K sub-offset
  const int koff2 = hi * 16;  // B-fragment K sub-offset
  v8f acc[4] = {};

  for (int kb = 0; kb < Cc; kb += 32) {
    v16h a;
    v8h* ap = (v8h*)&a;
    const _Float16* arow = X + (size_t)(row0 + nn) * Cc + kb;
    ap[0] = *(const v8h*)(arow + koff);
    ap[1] = *(const v8h*)(arow + 8 + koff);
#pragma unroll
    for (int t = 0; t < 4; ++t) {
      v16h b;
      v8h* bp = (v8h*)&b;
      const _Float16* bcol = W + (size_t)(col0 + t * 16 + nn) * Cc + kb + koff2;
      bp[0] = *(const v8h*)(bcol);
      bp[1] = *(const v8h*)(bcol + 8);
      acc[t] = WMMA(a, b, acc[t]);
    }
  }

#pragma unroll
  for (int t = 0; t < 4; ++t) {
    const int c3 = col0 + t * 16 + nn;
    const int s = c3 / Cc;
    const int rem = c3 - s * Cc;
    const int h = rem >> 6, d = rem & 63;
    _Float16* dst = (s == 0) ? Q : (s == 1) ? K : V;
#pragma unroll
    for (int r = 0; r < 8; ++r) {
      const int rg = row0 + r + hi * 8;
      const int bidx = rg >> 10, n = rg & 1023;
      dst[(size_t)((bidx * Hh + h) * Nn + n) * Dd + d] = (_Float16)acc[t][r];
    }
  }
}

// ---------------------------------------------------------------- Flash attention
// One block per (b*h, 128 query rows). Double-buffered async K/V staging.
constexpr int KVSZ = 32 * 72;  // padded 32x64 f16 tile (stride 72 halves)

__global__ void __launch_bounds__(256) attn_kernel(const _Float16* __restrict__ Q,
                                                   const _Float16* __restrict__ K,
                                                   const _Float16* __restrict__ V,
                                                   _Float16* __restrict__ O) {
  __shared__ __align__(16) _Float16 ldsK[2 * KVSZ];   // K blocks, row-major padded
  __shared__ __align__(16) _Float16 ldsV[2 * KVSZ];   // V blocks, row-major padded
  __shared__ __align__(16) _Float16 ldsVt[64 * 40];   // V transposed, stride 40
  __shared__ __align__(16) _Float16 ldsP[8 * 512];    // per-wave P scratch (16x32)

  const int tid = threadIdx.x;
  const int w = tid >> 5, lane = tid & 31;
  const int nn = lane & 15, hi = lane >> 4;
  const int bh = blockIdx.y;
  const int qrow0 = blockIdx.x * 128 + w * 16;
  const int rr = tid >> 3, c8 = (tid & 7) * 8;  // cooperative copy coords

  const _Float16* Qb = Q + (size_t)bh * Nn * Dd;
  const _Float16* Kb = K + (size_t)bh * Nn * Dd;
  const _Float16* Vb = V + (size_t)bh * Nn * Dd;

  // Q fragments (16 rows x 64 d), loaded once; A-layout.
  v16h qf0, qf1;
  {
    const _Float16* qrow = Qb + (size_t)(qrow0 + nn) * Dd;
    ((v8h*)&qf0)[0] = *(const v8h*)(qrow + hi * 16);
    ((v8h*)&qf0)[1] = *(const v8h*)(qrow + 8 + hi * 16);
    ((v8h*)&qf1)[0] = *(const v8h*)(qrow + 32 + hi * 16);
    ((v8h*)&qf1)[1] = *(const v8h*)(qrow + 40 + hi * 16);
  }

  v8f o0 = {}, o1 = {}, o2 = {}, o3 = {};
  float rmax[8], rsum[8];
#pragma unroll
  for (int r = 0; r < 8; ++r) { rmax[r] = -1e30f; rsum[r] = 0.0f; }

  _Float16* pbase = ldsP + w * 512;
  constexpr int NS = Nn / 32;

  // Prime the pipeline: async-stage block 0 into buffer 0 (one b128 chunk/thread).
  async_copy_b128(Kb + (size_t)rr * Dd + c8, ldsK + rr * 72 + c8);
  async_copy_b128(Vb + (size_t)rr * Dd + c8, ldsV + rr * 72 + c8);

  for (int jb = 0; jb < NS; ++jb) {
    const int cur = jb & 1;
    wait_async0();      // this wave's chunks of the current buffers are in LDS
    __syncthreads();    // publish everyone's chunks (and retire prior consumers)

    // Overlap: kick off next block's async staging into the other buffer.
    if (jb + 1 < NS) {
      const size_t gofs = (size_t)((jb + 1) * 32 + rr) * Dd + c8;
      const int lofs = (1 - cur) * KVSZ + rr * 72 + c8;
      async_copy_b128(Kb + gofs, ldsK + lofs);
      async_copy_b128(Vb + gofs, ldsV + lofs);
    }

    // Transpose V (row-major -> d-major) so PV B-fragments are contiguous.
    {
      v8h vv = *(const v8h*)(ldsV + cur * KVSZ + rr * 72 + c8);
#pragma unroll
      for (int i = 0; i < 8; ++i) ldsVt[(c8 + i) * 40 + rr] = vv[i];
    }
    __syncthreads();

    const _Float16* Kc = ldsK + cur * KVSZ;

    // S = Q * K^T  (two 16x16 tiles over the 32 keys)
    v8f s0 = {}, s1 = {};
    {
      v16h b;
      v8h* bp = (v8h*)&b;
      const _Float16* kr0 = Kc + (size_t)nn * 72 + hi * 16;
      bp[0] = *(const v8h*)(kr0);
      bp[1] = *(const v8h*)(kr0 + 8);
      s0 = WMMA(qf0, b, s0);
      bp[0] = *(const v8h*)(kr0 + 32);
      bp[1] = *(const v8h*)(kr0 + 40);
      s0 = WMMA(qf1, b, s0);
      const _Float16* kr1 = Kc + (size_t)(16 + nn) * 72 + hi * 16;
      bp[0] = *(const v8h*)(kr1);
      bp[1] = *(const v8h*)(kr1 + 8);
      s1 = WMMA(qf0, b, s1);
      bp[0] = *(const v8h*)(kr1 + 32);
      bp[1] = *(const v8h*)(kr1 + 40);
      s1 = WMMA(qf1, b, s1);
    }

    // Online softmax; C-layout rows are r + 8*hi, columns stripe over 16 lanes.
#pragma unroll
    for (int r = 0; r < 8; ++r) {
      float f0 = s0[r] * SCALE, f1 = s1[r] * SCALE;
      float mx = fmaxf(f0, f1);
      mx = fmaxf(mx, __shfl_xor(mx, 1));
      mx = fmaxf(mx, __shfl_xor(mx, 2));
      mx = fmaxf(mx, __shfl_xor(mx, 4));
      mx = fmaxf(mx, __shfl_xor(mx, 8));
      float mnew = fmaxf(rmax[r], mx);
      float p0 = __expf(f0 - mnew), p1 = __expf(f1 - mnew);
      float ps = p0 + p1;
      ps += __shfl_xor(ps, 1);
      ps += __shfl_xor(ps, 2);
      ps += __shfl_xor(ps, 4);
      ps += __shfl_xor(ps, 8);
      float alpha = __expf(rmax[r] - mnew);
      rsum[r] = rsum[r] * alpha + ps;
      rmax[r] = mnew;
      o0[r] *= alpha; o1[r] *= alpha; o2[r] *= alpha; o3[r] *= alpha;
      pbase[(r + hi * 8) * 32 + nn] = (_Float16)p0;
      pbase[(r + hi * 8) * 32 + 16 + nn] = (_Float16)p1;
    }

    // P back as an A-layout fragment (wave-local LDS round trip).
    v16h pf;
    ((v8h*)&pf)[0] = *(const v8h*)(pbase + nn * 32 + hi * 16);
    ((v8h*)&pf)[1] = *(const v8h*)(pbase + nn * 32 + 8 + hi * 16);

    // O += P * V  (4 tiles across d)
    {
      v16h b;
      v8h* bp = (v8h*)&b;
      const _Float16* vr;
      vr = ldsVt + (0 * 16 + nn) * 40 + hi * 16;
      bp[0] = *(const v8h*)vr; bp[1] = *(const v8h*)(vr + 8);
      o0 = WMMA(pf, b, o0);
      vr = ldsVt + (1 * 16 + nn) * 40 + hi * 16;
      bp[0] = *(const v8h*)vr; bp[1] = *(const v8h*)(vr + 8);
      o1 = WMMA(pf, b, o1);
      vr = ldsVt + (2 * 16 + nn) * 40 + hi * 16;
      bp[0] = *(const v8h*)vr; bp[1] = *(const v8h*)(vr + 8);
      o2 = WMMA(pf, b, o2);
      vr = ldsVt + (3 * 16 + nn) * 40 + hi * 16;
      bp[0] = *(const v8h*)vr; bp[1] = *(const v8h*)(vr + 8);
      o3 = WMMA(pf, b, o3);
    }
  }

  // Normalize and write f16 output in [B, N, C] layout (head-offset fused).
  const int b = bh / Hh, h = bh - b * Hh;
#pragma unroll
  for (int r = 0; r < 8; ++r) {
    float inv = 1.0f / rsum[r];
    const int n = qrow0 + r + hi * 8;
    size_t base = ((size_t)(b * Nn + n)) * Cc + h * Dd + nn;
    O[base + 0]  = (_Float16)(o0[r] * inv);
    O[base + 16] = (_Float16)(o1[r] * inv);
    O[base + 32] = (_Float16)(o2[r] * inv);
    O[base + 48] = (_Float16)(o3[r] * inv);
  }
}

// ---------------------------------------------------------------- Proj GEMM + bias
__global__ void __launch_bounds__(256) proj_gemm(const _Float16* __restrict__ X,
                                                 const _Float16* __restrict__ W,
                                                 const float* __restrict__ bias,
                                                 float* __restrict__ Out) {
  const int tid = threadIdx.x;
  const int w = tid >> 5, lane = tid & 31;
  const int nn = lane & 15, hi = lane >> 4;
  const int row0 = blockIdx.y * 128 + w * 16;
  const int col0 = blockIdx.x * 64;
  const int koff = hi * 16;
  const int koff2 = hi * 16;
  v8f acc[4] = {};

  for (int kb = 0; kb < Cc; kb += 32) {
    v16h a;
    v8h* ap = (v8h*)&a;
    const _Float16* arow = X + (size_t)(row0 + nn) * Cc + kb;
    ap[0] = *(const v8h*)(arow + koff);
    ap[1] = *(const v8h*)(arow + 8 + koff);
#pragma unroll
    for (int t = 0; t < 4; ++t) {
      v16h b;
      v8h* bp = (v8h*)&b;
      const _Float16* bcol = W + (size_t)(col0 + t * 16 + nn) * Cc + kb + koff2;
      bp[0] = *(const v8h*)(bcol);
      bp[1] = *(const v8h*)(bcol + 8);
      acc[t] = WMMA(a, b, acc[t]);
    }
  }

#pragma unroll
  for (int t = 0; t < 4; ++t) {
    const int col = col0 + t * 16 + nn;
    const float bv = bias[col];
#pragma unroll
    for (int r = 0; r < 8; ++r) {
      const int rg = row0 + r + hi * 8;
      Out[(size_t)rg * Cc + col] = acc[t][r] + bv;
    }
  }
}

// ---------------------------------------------------------------- launch
extern "C" void kernel_launch(void* const* d_in, const int* in_sizes, int n_in,
                              void* d_out, int out_size, void* d_ws, size_t ws_size,
                              hipStream_t stream) {
  const float* x = (const float*)d_in[0];      // [8,1024,768]
  const float* wqkv = (const float*)d_in[1];   // [2304,768]
  const float* wproj = (const float*)d_in[2];  // [768,768]
  const float* bproj = (const float*)d_in[3];  // [768]
  float* out = (float*)d_out;                  // [8,1024,768]

  char* ws = (char*)d_ws;
  size_t off = 0;
  auto alloc = [&](size_t bytes) -> void* {
    void* p = ws + off;
    off += (bytes + 255) & ~(size_t)255;
    return p;
  };
  _Float16* xh = (_Float16*)alloc((size_t)Bb * Nn * Cc * 2);
  _Float16* wqkvh = (_Float16*)alloc((size_t)C3 * Cc * 2);
  _Float16* wprojh = (_Float16*)alloc((size_t)Cc * Cc * 2);
  _Float16* qb = (_Float16*)alloc((size_t)Bb * Hh * Nn * Dd * 2);
  _Float16* kb = (_Float16*)alloc((size_t)Bb * Hh * Nn * Dd * 2);
  _Float16* vb = (_Float16*)alloc((size_t)Bb * Hh * Nn * Dd * 2);
  _Float16* oh = (_Float16*)alloc((size_t)Bb * Nn * Cc * 2);

  cvt_f32_f16<<<(Bb * Nn * Cc) / 1024, 256, 0, stream>>>(x, xh);
  cvt_f32_f16<<<(C3 * Cc) / 1024, 256, 0, stream>>>(wqkv, wqkvh);
  cvt_f32_f16<<<(Cc * Cc) / 1024, 256, 0, stream>>>(wproj, wprojh);

  qkv_gemm<<<dim3(C3 / 64, (Bb * Nn) / 128), 256, 0, stream>>>(xh, wqkvh, qb, kb, vb);
  attn_kernel<<<dim3(Nn / 128, Bb * Hh), 256, 0, stream>>>(qb, kb, vb, oh);
  proj_gemm<<<dim3(Cc / 64, (Bb * Nn) / 128), 256, 0, stream>>>(oh, wprojh, bproj, out);
}
